// DInfoNCELoss_24721831755975
// MI455X (gfx1250) — matrix operand order repaired and checked by reading, not compiled, not run on hardware
//
#include <hip/hip_runtime.h>

#define BDIM 16384
#define CDIM 2048
#define DDIM 128
#define KSPLIT 8
#define MSPLIT 64

typedef __bf16 v16bf __attribute__((ext_vector_type(16)));
typedef __bf16 v8bf  __attribute__((ext_vector_type(8)));
typedef float  v8f   __attribute__((ext_vector_type(8)));
typedef unsigned int u32x4 __attribute__((ext_vector_type(4)));
typedef unsigned int u32x8 __attribute__((ext_vector_type(8)));

union BfVec { v16bf v; v8bf h[2]; };
union BfPack { __bf16 b[2]; unsigned u; };

static __device__ __forceinline__ v8f wmma_bf16f32(v16bf a, v16bf b, v8f c) {
  // D = A*B + C, 16x16x32 bf16 -> f32 accumulate
  return __builtin_amdgcn_wmma_f32_16x16x32_bf16(false, a, false, b, (short)0, c, false, false);
}

static __device__ __forceinline__ v8f v8f_zero() {
  v8f z = {0.f, 0.f, 0.f, 0.f, 0.f, 0.f, 0.f, 0.f};
  return z;
}

// ---------------------------------------------------------------------------
// TDM: DMA a 2D bf16 tile (rows x rowlen elems, global row stride
// gstride elems) from global memory into contiguous LDS rows.
// Descriptor per CDNA5 ISA 8.3/8.4: count=1, type=2 (image), data_size=2B,
// tensor dims set huge so no OOB clipping; groups 2/3 zero (tile_dim2..4=0).
// ---------------------------------------------------------------------------
static __device__ __forceinline__ void tdm_load_2d_bf16(
    const __bf16* gsrc, unsigned lds_off, unsigned rows, unsigned rowlen,
    unsigned gstride) {
  const unsigned long long ga = (unsigned long long)(uintptr_t)gsrc;
  u32x4 g0;
  g0[0] = 1u;                                   // count=1 (user descriptor)
  g0[1] = lds_off;                              // lds_addr (bytes)
  g0[2] = (unsigned)(ga & 0xffffffffu);         // global_addr[95:64]
  g0[3] = (unsigned)((ga >> 32) & 0x01ffffffu)  // global_addr[120:96]
          | (2u << 30);                         // type=2
  u32x8 g1;
  g1[0] = (1u << 16);                // workgroup_mask=0, data_size=1 (2 bytes)
  g1[1] = 0u;                        // abar=0, tensor_dim0 low16 (td0=1<<30)
  g1[2] = 0x4000u;                   // td0 hi16 | td1 low16 (td1=1<<30)
  g1[3] = 0x4000u | (rowlen << 16);  // td1 hi16 | tile_dim0
  g1[4] = rows;                      // tile_dim1 | tile_dim2=0
  g1[5] = gstride;                   // tensor_dim0_stride low32
  g1[6] = 0u;                        // stride hi16 | tensor_dim1_stride low
  g1[7] = 0u;
  const u32x4 gz = {0u, 0u, 0u, 0u};
  asm volatile("tensor_load_to_lds %0, %1, %2, %3"
               :: "s"(g0), "s"(g1), "s"(gz), "s"(gz)
               : "memory");
}

// ---------------------------------------------------------------------------
// mask_count partials: maskp[s][c] = sum over b-slice s of targets[b][c]
// ---------------------------------------------------------------------------
__global__ __launch_bounds__(256) void k_maskpart(const float* __restrict__ targets,
                                                  float* __restrict__ maskp) {
  const int c = blockIdx.x * 256 + threadIdx.x;
  const int sp = blockIdx.y;
  const int b0 = sp * (BDIM / MSPLIT);
  float s = 0.f;
  for (int b = 0; b < BDIM / MSPLIT; ++b)
    s += targets[(size_t)(b0 + b) * CDIM + c];
  maskp[(size_t)sp * CDIM + c] = s;
}

// ---------------------------------------------------------------------------
// Phase 1: pool_part[ks][c][d] = sum_b T[b,c]*(Ohi+Olo)[b,d]  (split bf16)
//          cnt_part[ks][c][d]  = sum_b T[b,c]*(O[b,d]!=0)     (exact)
// Manual staging (needs f32->bf16 conversion in flight) + prefetch.
// ---------------------------------------------------------------------------
__global__ __launch_bounds__(256) void k_gemm_pool(
    const float* __restrict__ targets, const float* __restrict__ output,
    float* __restrict__ pool_part, float* __restrict__ cnt_part) {
  __shared__ __align__(32) __bf16 ldsA[128 * 32];   // [c_local][b_local]
  __shared__ __align__(32) __bf16 ldsHi[128 * 32];  // [d][b_local]
  __shared__ __align__(32) __bf16 ldsLo[128 * 32];
  __shared__ __align__(32) __bf16 ldsNz[128 * 32];

  const int cbase = blockIdx.x * 128;
  const int ks = blockIdx.y;
  const int t = threadIdx.x;
  const int wave = t >> 5;
  const int lane = t & 31;
  const int Lm = lane & 15;
  const int Lh = lane >> 4;

  v8f accP[8], accC[8];
#pragma unroll
  for (int i = 0; i < 8; ++i) { accP[i] = v8f_zero(); accC[i] = v8f_zero(); }

  const int b0 = ks * (BDIM / KSPLIT);
  for (int kb = 0; kb < BDIM / KSPLIT; kb += 32) {
    const bool more = (kb + 32) < (BDIM / KSPLIT);
    // stage A: targets tile, two b-rows packed per u32 LDS store
#pragma unroll
    for (int e = t; e < 16 * 128; e += 256) {
      const int cl = e & 127, bp = e >> 7;
      const size_t g = (size_t)(b0 + kb + 2 * bp) * CDIM + cbase + cl;
      BfPack pk;
      pk.b[0] = (__bf16)targets[g];         // targets exactly 0/1 in bf16
      pk.b[1] = (__bf16)targets[g + CDIM];
      *(unsigned*)&ldsA[cl * 32 + 2 * bp] = pk.u;
      if (more && (cl & 31) == 0)
        __builtin_prefetch(&targets[g + (size_t)32 * CDIM], 0, 1);
    }
    // stage B: output tile -> hi/lo/nz, packed stores
#pragma unroll
    for (int e = t; e < 16 * 128; e += 256) {
      const int dl = e & 127, bp = e >> 7;
      const size_t g = (size_t)(b0 + kb + 2 * bp) * DDIM + dl;
      const float v0 = output[g];
      const float v1 = output[g + DDIM];
      const __bf16 h0 = (__bf16)v0, h1 = (__bf16)v1;
      BfPack hi, lo, nz;
      hi.b[0] = h0; hi.b[1] = h1;
      lo.b[0] = (__bf16)(v0 - (float)h0); lo.b[1] = (__bf16)(v1 - (float)h1);
      nz.b[0] = (v0 != 0.0f) ? (__bf16)1.0f : (__bf16)0.0f;
      nz.b[1] = (v1 != 0.0f) ? (__bf16)1.0f : (__bf16)0.0f;
      const int o = dl * 32 + 2 * bp;
      *(unsigned*)&ldsHi[o] = hi.u;
      *(unsigned*)&ldsLo[o] = lo.u;
      *(unsigned*)&ldsNz[o] = nz.u;
      if (more && (dl & 31) == 0)
        __builtin_prefetch(&output[g + (size_t)32 * DDIM], 0, 1);
    }
    __syncthreads();

    BfVec a;  // A 16x32 bf16: lane holds row M=Lm, K = 16*Lh..+15 contiguous
    const int arow = (wave * 16 + Lm) * 32;
    a.h[0] = *(const v8bf*)&ldsA[arow + 16 * Lh];
    a.h[1] = *(const v8bf*)&ldsA[arow + 16 * Lh + 8];

    // software-pipelined B fragments (lane = N, K = 16*Lh..+15 contiguous)
    const int bo0 = Lm * 32 + 16 * Lh;
    v16bf bnz = *(const v16bf*)&ldsNz[bo0];
    v16bf bhi = *(const v16bf*)&ldsHi[bo0];
    v16bf blo = *(const v16bf*)&ldsLo[bo0];
#pragma unroll
    for (int nt = 0; nt < 8; ++nt) {
      v16bf nnz, nhi, nlo;
      if (nt < 7) {
        const int bo = ((nt + 1) * 16 + Lm) * 32 + 16 * Lh;
        nnz = *(const v16bf*)&ldsNz[bo];
        nhi = *(const v16bf*)&ldsHi[bo];
        nlo = *(const v16bf*)&ldsLo[bo];
      }
      accC[nt] = wmma_bf16f32(a.v, bnz, accC[nt]);
      accP[nt] = wmma_bf16f32(a.v, bhi, accP[nt]);
      accP[nt] = wmma_bf16f32(a.v, blo, accP[nt]);
      if (nt < 7) { bnz = nnz; bhi = nhi; blo = nlo; }
    }
    __syncthreads();
  }

#pragma unroll
  for (int nt = 0; nt < 8; ++nt) {
#pragma unroll
    for (int r = 0; r < 8; ++r) {
      const int M = wave * 16 + 8 * Lh + r;  // C/D layout: M = r + 8*(lane>=16)
      const int N = nt * 16 + Lm;            // N = lane & 15
      const size_t idx = ((size_t)ks * CDIM + cbase + M) * DDIM + N;
      pool_part[idx] = accP[nt][r];
      cnt_part[idx] = accC[nt][r];
    }
  }
}

// ---------------------------------------------------------------------------
// Phase 2: reduce K-split partials, pooled = num/(cnt+1e-9), L2-normalize rows
// of pooled (blocks [0,C)) and class_emb (blocks [C,2C)); emit bf16 hi/lo.
// ---------------------------------------------------------------------------
__global__ __launch_bounds__(128) void k_finalize(
    const float* __restrict__ class_emb, const float* __restrict__ pool_part,
    const float* __restrict__ cnt_part, __bf16* __restrict__ pnh,
    __bf16* __restrict__ pnl, __bf16* __restrict__ enh, __bf16* __restrict__ enl) {
  __shared__ float red[128];
  const int row = blockIdx.x;
  const int d = threadIdx.x;
  const bool isEmb = row >= CDIM;
  const int c = isEmb ? row - CDIM : row;
  float val;
  if (!isEmb) {
    float ps = 0.f, cs = 0.f;
#pragma unroll
    for (int s = 0; s < KSPLIT; ++s) {
      const size_t idx = ((size_t)s * CDIM + c) * DDIM + d;
      ps += pool_part[idx];
      cs += cnt_part[idx];
    }
    val = ps / (cs + 1e-9f);
  } else {
    val = class_emb[(size_t)c * DDIM + d];
  }
  red[d] = val * val;
  __syncthreads();
  for (int s = 64; s > 0; s >>= 1) {
    if (d < s) red[d] += red[d + s];
    __syncthreads();
  }
  const float nv = val * rsqrtf(red[0] + 1e-12f);
  const __bf16 h = (__bf16)nv;
  const __bf16 l = (__bf16)(nv - (float)h);
  const size_t o = (size_t)c * DDIM + d;
  if (!isEmb) { pnh[o] = h; pnl[o] = l; }
  else        { enh[o] = h; enl[o] = l; }
}

// ---------------------------------------------------------------------------
// Phase 3: cosm = pn@en^T, simm = en@en^T  (split-bf16, drop lo*lo term)
// Tiles are pure 2D bf16 copies -> staged by the Tensor Data Mover from
// wave 0 (TDM ignores EXEC), s_wait_tensorcnt 0, then block barrier.
// ---------------------------------------------------------------------------
__global__ __launch_bounds__(256) void k_gemm_sim(
    const __bf16* __restrict__ pnh, const __bf16* __restrict__ pnl,
    const __bf16* __restrict__ enh, const __bf16* __restrict__ enl,
    float* __restrict__ cosm, float* __restrict__ simm) {
  __shared__ __align__(32) __bf16 Aph[128 * 32];
  __shared__ __align__(32) __bf16 Apl[128 * 32];
  __shared__ __align__(32) __bf16 Aeh[128 * 32];
  __shared__ __align__(32) __bf16 Ael[128 * 32];
  __shared__ __align__(32) __bf16 Beh[128 * 32];
  __shared__ __align__(32) __bf16 Bel[128 * 32];

  const int cb = blockIdx.x * 128;
  const int nb = blockIdx.y * 128;
  const int t = threadIdx.x, wave = t >> 5, lane = t & 31;
  const int Lm = lane & 15, Lh = lane >> 4;

  v8f aC[8], aS[8];
#pragma unroll
  for (int i = 0; i < 8; ++i) { aC[i] = v8f_zero(); aS[i] = v8f_zero(); }

  for (int kd = 0; kd < DDIM; kd += 32) {
    if (wave == 0) {
      tdm_load_2d_bf16(&pnh[(size_t)cb * DDIM + kd],
                       (unsigned)(uintptr_t)(void*)&Aph[0], 128, 32, DDIM);
      tdm_load_2d_bf16(&pnl[(size_t)cb * DDIM + kd],
                       (unsigned)(uintptr_t)(void*)&Apl[0], 128, 32, DDIM);
      tdm_load_2d_bf16(&enh[(size_t)cb * DDIM + kd],
                       (unsigned)(uintptr_t)(void*)&Aeh[0], 128, 32, DDIM);
      tdm_load_2d_bf16(&enl[(size_t)cb * DDIM + kd],
                       (unsigned)(uintptr_t)(void*)&Ael[0], 128, 32, DDIM);
      tdm_load_2d_bf16(&enh[(size_t)nb * DDIM + kd],
                       (unsigned)(uintptr_t)(void*)&Beh[0], 128, 32, DDIM);
      tdm_load_2d_bf16(&enl[(size_t)nb * DDIM + kd],
                       (unsigned)(uintptr_t)(void*)&Bel[0], 128, 32, DDIM);
      __builtin_amdgcn_s_wait_tensorcnt(0);
    }
    __syncthreads();

    const int arow = (wave * 16 + Lm) * 32;
    BfVec aph, apl, aeh, ael;
    aph.h[0] = *(const v8bf*)&Aph[arow + 16 * Lh];
    aph.h[1] = *(const v8bf*)&Aph[arow + 16 * Lh + 8];
    apl.h[0] = *(const v8bf*)&Apl[arow + 16 * Lh];
    apl.h[1] = *(const v8bf*)&Apl[arow + 16 * Lh + 8];
    aeh.h[0] = *(const v8bf*)&Aeh[arow + 16 * Lh];
    aeh.h[1] = *(const v8bf*)&Aeh[arow + 16 * Lh + 8];
    ael.h[0] = *(const v8bf*)&Ael[arow + 16 * Lh];
    ael.h[1] = *(const v8bf*)&Ael[arow + 16 * Lh + 8];

    // software-pipelined B fragments
    const int bo0 = Lm * 32 + 16 * Lh;
    v16bf beh = *(const v16bf*)&Beh[bo0];
    v16bf bel = *(const v16bf*)&Bel[bo0];
#pragma unroll
    for (int nt = 0; nt < 8; ++nt) {
      v16bf neh, nel;
      if (nt < 7) {
        const int bo = ((nt + 1) * 16 + Lm) * 32 + 16 * Lh;
        neh = *(const v16bf*)&Beh[bo];
        nel = *(const v16bf*)&Bel[bo];
      }
      aC[nt] = wmma_bf16f32(aph.v, beh, aC[nt]);
      aC[nt] = wmma_bf16f32(aph.v, bel, aC[nt]);
      aC[nt] = wmma_bf16f32(apl.v, beh, aC[nt]);
      aS[nt] = wmma_bf16f32(aeh.v, beh, aS[nt]);
      aS[nt] = wmma_bf16f32(aeh.v, bel, aS[nt]);
      aS[nt] = wmma_bf16f32(ael.v, beh, aS[nt]);
      if (nt < 7) { beh = neh; bel = nel; }
    }
    __syncthreads();
  }

#pragma unroll
  for (int nt = 0; nt < 8; ++nt) {
#pragma unroll
    for (int r = 0; r < 8; ++r) {
      const int M = wave * 16 + 8 * Lh + r;
      const int N = nt * 16 + Lm;
      const size_t idx = (size_t)(cb + M) * CDIM + nb + N;
      cosm[idx] = aC[nt][r];
      simm[idx] = aS[nt][r];
    }
  }
}

// ---------------------------------------------------------------------------
// Phase 4: per row c — bitonic argsort of sim row (stable tie-break by index),
// rank = inverse permutation, logits = cos/(T*log(rank+2)), log-softmax diag.
// Sim row pulled into LDS with async global->LDS b128 copies (ASYNCcnt).
// ---------------------------------------------------------------------------
__global__ __launch_bounds__(256) void k_row_loss(
    const float* __restrict__ cosm, const float* __restrict__ simm,
    const float* __restrict__ temperature, float* __restrict__ plp) {
  __shared__ __align__(16) float sv[CDIM];
  __shared__ int si[CDIM];
  __shared__ int rk[CDIM];
  __shared__ float red[256];
  const int c = blockIdx.x, t = threadIdx.x;
  const float* srow = simm + (size_t)c * CDIM;
  // async copy sim row -> sv (16B chunks, per-lane LDS dest)
  for (int ch = t; ch < CDIM / 4; ch += 256) {
    const unsigned lds = (unsigned)(uintptr_t)(void*)&sv[ch * 4];
    const unsigned long long ga = (unsigned long long)(uintptr_t)(srow + ch * 4);
    asm volatile("global_load_async_to_lds_b128 %0, %1, off"
                 :: "v"(lds), "v"(ga) : "memory");
  }
  for (int i = t; i < CDIM; i += 256) si[i] = i;
  asm volatile("s_wait_asynccnt 0x0" ::: "memory");
  __syncthreads();
  for (int k = 2; k <= CDIM; k <<= 1) {
    for (int j = k >> 1; j > 0; j >>= 1) {
      for (int i = t; i < CDIM; i += 256) {
        const int l = i ^ j;
        if (l > i) {
          const bool up = ((i & k) == 0);
          const float v0 = sv[i], v1 = sv[l];
          const int i0 = si[i], i1 = si[l];
          const bool gt = (v0 > v1) || (v0 == v1 && i0 > i1);
          if (gt == up) { sv[i] = v1; sv[l] = v0; si[i] = i1; si[l] = i0; }
        }
      }
      __syncthreads();
    }
  }
  for (int p = t; p < CDIM; p += 256) rk[si[p]] = p;  // inverse permutation
  __syncthreads();
  const float T = temperature[0];
  const float* crow = cosm + (size_t)c * CDIM;
  float mx = -3.4e38f;
  for (int i = t; i < CDIM; i += 256) {
    const float lg = crow[i] / (T * logf((float)rk[i] + 2.0f));
    sv[i] = lg;  // reuse as logits
    mx = fmaxf(mx, lg);
  }
  red[t] = mx;
  __syncthreads();
  for (int s = 128; s > 0; s >>= 1) {
    if (t < s) red[t] = fmaxf(red[t], red[t + s]);
    __syncthreads();
  }
  const float mxv = red[0];
  __syncthreads();
  float sum = 0.f;
  for (int i = t; i < CDIM; i += 256) sum += expf(sv[i] - mxv);
  red[t] = sum;
  __syncthreads();
  for (int s = 128; s > 0; s >>= 1) {
    if (t < s) red[t] += red[t + s];
    __syncthreads();
  }
  if (t == 0) plp[c] = sv[c] - (mxv + logf(red[0]));
}

// ---------------------------------------------------------------------------
// Final: masked mean of -pos_log_p (deterministic single-block reduction)
// ---------------------------------------------------------------------------
__global__ __launch_bounds__(256) void k_reduce(const float* __restrict__ plp,
                                                const float* __restrict__ maskp,
                                                float* __restrict__ out) {
  __shared__ float rs[256];
  __shared__ float rc[256];
  const int t = threadIdx.x;
  float s = 0.f, n = 0.f;
  for (int c = t; c < CDIM; c += 256) {
    float mc = 0.f;
    for (int sp = 0; sp < MSPLIT; ++sp) mc += maskp[(size_t)sp * CDIM + c];
    if (mc > 0.0f) { s += -plp[c]; n += 1.0f; }
  }
  rs[t] = s; rc[t] = n;
  __syncthreads();
  for (int k = 128; k > 0; k >>= 1) {
    if (t < k) { rs[t] += rs[t + k]; rc[t] += rc[t + k]; }
    __syncthreads();
  }
  if (t == 0) out[0] = rs[0] / fmaxf(rc[0], 1.0f);
}

extern "C" void kernel_launch(void* const* d_in, const int* in_sizes, int n_in,
                              void* d_out, int out_size, void* d_ws, size_t ws_size,
                              hipStream_t stream) {
  (void)in_sizes; (void)n_in; (void)out_size; (void)ws_size;
  const float* output = (const float*)d_in[0];
  const float* class_emb = (const float*)d_in[1];
  const float* targets = (const float*)d_in[2];
  const float* temperature = (const float*)d_in[3];

  char* p = (char*)d_ws;
  auto alloc = [&](size_t bytes) -> void* {
    void* r = (void*)p;
    p += (bytes + 255) & ~(size_t)255;
    return r;
  };
  float* pool_part = (float*)alloc((size_t)KSPLIT * CDIM * DDIM * sizeof(float));
  float* cnt_part = (float*)alloc((size_t)KSPLIT * CDIM * DDIM * sizeof(float));
  float* maskp = (float*)alloc((size_t)MSPLIT * CDIM * sizeof(float));
  __bf16* pnh = (__bf16*)alloc((size_t)CDIM * DDIM * sizeof(__bf16));
  __bf16* pnl = (__bf16*)alloc((size_t)CDIM * DDIM * sizeof(__bf16));
  __bf16* enh = (__bf16*)alloc((size_t)CDIM * DDIM * sizeof(__bf16));
  __bf16* enl = (__bf16*)alloc((size_t)CDIM * DDIM * sizeof(__bf16));
  float* cosm = (float*)alloc((size_t)CDIM * CDIM * sizeof(float));
  float* simm = (float*)alloc((size_t)CDIM * CDIM * sizeof(float));
  float* plp = (float*)alloc((size_t)CDIM * sizeof(float));

  k_maskpart<<<dim3(CDIM / 256, MSPLIT), 256, 0, stream>>>(targets, maskp);
  k_gemm_pool<<<dim3(CDIM / 128, KSPLIT), 256, 0, stream>>>(targets, output,
                                                            pool_part, cnt_part);
  k_finalize<<<dim3(2 * CDIM), 128, 0, stream>>>(class_emb, pool_part, cnt_part,
                                                 pnh, pnl, enh, enl);
  k_gemm_sim<<<dim3(CDIM / 128, CDIM / 128), 256, 0, stream>>>(pnh, pnl, enh, enl,
                                                               cosm, simm);
  k_row_loss<<<dim3(CDIM), 256, 0, stream>>>(cosm, simm, temperature, plp);
  k_reduce<<<dim3(1), 256, 0, stream>>>(plp, maskp, (float*)d_out);
}
